// MTGNN_76811195121818
// MI455X (gfx1250) — matrix operand order repaired and checked by reading, not compiled, not run on hardware
//
#include <hip/hip_runtime.h>
#include <hip/hip_bf16.h>

// ---------------- problem constants ----------------
#define NN      5000      // nodes
#define NP      5008      // nodes padded to 16
#define NT      313       // 16-col tiles (313*16 = 5008)
#define NSPLIT  8         // column splits for the sim+topk kernel
#define DIMF    40        // node embedding dim
#define KP      64        // K padded to 2x32 for wmma f16 k=32
#define BB      8         // batch
#define CIN     16
#define TT      12
#define TP      6         // T - 6
#define FEAT    768       // B*CIN*TP = 8*16*6
#define KTOP    20
#define NNZ     (NN*KTOP) // 100000
#define ALPHA   0.05f

typedef _Float16 v16h __attribute__((ext_vector_type(16)));
typedef _Float16 v8h  __attribute__((ext_vector_type(8)));
typedef float    v8f  __attribute__((ext_vector_type(8)));

// branch-free transcendentals -> single hardware v_exp_f32 each (no libm exec-mask dance)
__device__ __forceinline__ float fast_tanh(float x) {
  // tanh(x) = 1 - 2/(1+exp(2x)); exp overflow/underflow saturate correctly to +/-1
  return 1.0f - 2.0f / (1.0f + __expf(2.0f * x));
}
__device__ __forceinline__ float fast_sigmoid(float x) {
  return 1.0f / (1.0f + __expf(-x));
}

__device__ __forceinline__ v8f wmma_f16(v16h a, v16h b, v8f c) {
  // D = A(16x32 f16) x B(32x16 f16) + C(16x16 f32)
  return __builtin_amdgcn_wmma_f32_16x16x32_f16(
      /*neg_a=*/false, a, /*neg_b=*/false, b,
      /*c_mod=*/(short)0, c, /*reuse_a=*/false, /*reuse_b=*/false);
}

// A fragment: rows [m0..m0+15], k-step s (K 32s..32s+31).
// lane<16 : row=m0+lane,    elems 0-7 = K base+0..7,  8-15 = K base+16..23, base=32s
// lane>=16: row=m0+lane-16, elems 0-7 = K base+8..15, 8-15 = K base+24..31
__device__ __forceinline__ v16h load_afrag(const _Float16* nv, int m0, int s, int lane) {
  int row = m0 + (lane & 15);
  int kb  = s * 32 + ((lane < 16) ? 0 : 8);
  const _Float16* p = nv + (size_t)row * KP + kb;
  v8h lo = *(const v8h*)p;
  v8h hi = *(const v8h*)(p + 16);
  v16h r;
#pragma unroll
  for (int i = 0; i < 8; ++i) { r[i] = lo[i]; r[8 + i] = hi[i]; }
  return r;
}

// B fragment: B[k][n] = nv[n][k] for cols [n0..n0+15], k-step s.
// lane col = n0 + (lane&15); elems 0-7 = K base+h8..+7, 8-15 = K base+16+h8..+7, h8 = lane<16 ? 0 : 8.
__device__ __forceinline__ v16h load_bfrag(const _Float16* nv, int n0, int s, int lane) {
  int col = n0 + (lane & 15);
  int kb  = s * 32 + ((lane < 16) ? 0 : 8);
  const _Float16* p = nv + (size_t)col * KP + kb;
  v8h lo = *(const v8h*)p;
  v8h hi = *(const v8h*)(p + 16);
  v16h r;
#pragma unroll
  for (int i = 0; i < 8; ++i) { r[i] = lo[i]; r[8 + i] = hi[i]; }
  return r;
}

// ---------------- kernel 1: node vectors nv = tanh(3*(emb[idx] @ W^T + b)), f16, padded ----
__global__ __launch_bounds__(64) void k_node_vecs(const float* __restrict__ emb,
                                                  const int* __restrict__ idxmap,
                                                  const float* __restrict__ W,
                                                  const float* __restrict__ bias,
                                                  _Float16* __restrict__ nv) {
  int n = blockIdx.x;          // 0..NP-1
  int j = threadIdx.x;         // 0..63
  float v = 0.f;
  if (n < NN && j < DIMF) {
    int src = idxmap[n];
    const float* e = emb + (size_t)src * DIMF;
    const float* w = W + (size_t)j * DIMF;
    float s = bias[j];
#pragma unroll 8
    for (int k = 0; k < DIMF; ++k) s += e[k] * w[k];
    v = fast_tanh(3.0f * s);
  }
  nv[(size_t)n * KP + j] = (_Float16)v;
}

// ---------------- kernel 2: WMMA similarity + fused per-row top-20 -------------------------
// relu(tanh(3a)) is monotone in a, so top-k runs on RAW wmma outputs; tanh is deferred to
// the merge kernel (20 survivors/row instead of N^2 activations).
// grid (NT row-blocks, NSPLIT col-splits), block = 1 wave (32).
__global__ __launch_bounds__(32) void k_simtopk(const _Float16* __restrict__ nv1,
                                                const _Float16* __restrict__ nv2,
                                                float* __restrict__ cand_val,
                                                int* __restrict__ cand_idx) {
  __shared__ float sT[16][17];
  __shared__ float lval[16][KTOP];
  __shared__ int   lidx[16][KTOP];
  __shared__ float lmin[16];
  __shared__ int   lminp[16];

  const int lane  = threadIdx.x;
  const int mb    = blockIdx.x;
  const int split = blockIdx.y;
  const int m0    = mb * 16;

  for (int i = lane; i < 16 * KTOP; i += 32) { lval[i / KTOP][i % KTOP] = 0.f; lidx[i / KTOP][i % KTOP] = 0; }
  if (lane < 16) { lmin[lane] = 0.f; lminp[lane] = 0; }
  __syncthreads();

  // A operands for this 16-row block (loaded once)
  v16h a1lo = load_afrag(nv1, m0, 0, lane), a1hi = load_afrag(nv1, m0, 1, lane);
  v16h a2lo = load_afrag(nv2, m0, 0, lane), a2hi = load_afrag(nv2, m0, 1, lane);

  const int t0 = (split * NT) / NSPLIT;
  const int t1 = ((split + 1) * NT) / NSPLIT;
  const int colL  = lane & 15;
  const int rbase = (lane < 16) ? 0 : 8;

  for (int tl = t0; tl < t1; ++tl) {
    int n0 = tl * 16;
    v16h b2lo = load_bfrag(nv2, n0, 0, lane), b2hi = load_bfrag(nv2, n0, 1, lane);
    v16h b1lo = load_bfrag(nv1, n0, 0, lane), b1hi = load_bfrag(nv1, n0, 1, lane);
    v8f c1 = {}; v8f c2 = {};
    c1 = wmma_f16(a1lo, b2lo, c1);   // nv1 . nv2^T
    c1 = wmma_f16(a1hi, b2hi, c1);
    c2 = wmma_f16(a2lo, b1lo, c2);   // nv2 . nv1^T
    c2 = wmma_f16(a2hi, b1hi, c2);
#pragma unroll
    for (int r = 0; r < 8; ++r) {
      sT[rbase + r][colL] = c1[r] - c2[r];   // raw similarity, no activation here
    }
    __syncthreads();
    if (lane < 16) {                       // each lane owns one row's top-20 list
      int r = lane;
      float mn = lmin[r]; int mp = lminp[r];
#pragma unroll
      for (int j = 0; j < 16; ++j) {
        float v = sT[r][j];
        int gc = n0 + j;
        if (v > mn && gc < NN) {           // mn >= 0 always => only positives enter
          lval[r][mp] = v; lidx[r][mp] = gc;
          mn = lval[r][0]; mp = 0;
          for (int q = 1; q < KTOP; ++q) { float lv = lval[r][q]; if (lv < mn) { mn = lv; mp = q; } }
        }
      }
      lmin[r] = mn; lminp[r] = mp;
    }
    __syncthreads();
  }
  if (lane < 16) {
    int gr = m0 + lane;
    if (gr < NN) {
      size_t base = ((size_t)gr * NSPLIT + split) * KTOP;
      for (int q = 0; q < KTOP; ++q) { cand_val[base + q] = lval[lane][q]; cand_idx[base + q] = lidx[lane][q]; }
    }
  }
}

// ---------------- kernel 3: merge 8x20 candidates -> final top-20 per row ------------------
// Applies the deferred activation relu(tanh(3s)) to the 20 survivors only.
__global__ __launch_bounds__(32) void k_merge(const float* __restrict__ cand_val,
                                              const int* __restrict__ cand_idx,
                                              float* __restrict__ adj_val,
                                              int* __restrict__ adj_idx,
                                              float* __restrict__ rowsum,
                                              int* __restrict__ colcnt) {
  __shared__ float cv[NSPLIT * KTOP];
  __shared__ int   ci[NSPLIT * KTOP];
  __shared__ float bestv[32];
  __shared__ int   bestp[32];
  __shared__ float selv[KTOP];
  __shared__ int   seli[KTOP];
  const int row  = blockIdx.x;
  const int lane = threadIdx.x;
  for (int e = lane; e < NSPLIT * KTOP; e += 32) {
    cv[e] = cand_val[(size_t)row * NSPLIT * KTOP + e];
    ci[e] = cand_idx[(size_t)row * NSPLIT * KTOP + e];
  }
  __syncthreads();
  for (int k = 0; k < KTOP; ++k) {
    float bv = -1.f; int bp = -1;
#pragma unroll
    for (int q = 0; q < 5; ++q) {
      int e = lane * 5 + q;
      if (cv[e] > bv) { bv = cv[e]; bp = e; }
    }
    bestv[lane] = bv; bestp[lane] = bp;
    __syncthreads();
    if (lane == 0) {
      float gv = -1.f; int gp = 0;
      for (int q = 0; q < 32; ++q) if (bestv[q] > gv) { gv = bestv[q]; gp = bestp[q]; }
      float s = cv[gp];
      selv[k] = (s > 0.f) ? fast_tanh(3.0f * s) : 0.f;   // deferred activation
      seli[k] = ci[gp];
      cv[gp] = -2.f;
    }
    __syncthreads();
  }
  if (lane < KTOP) {
    adj_val[(size_t)row * KTOP + lane] = selv[lane];
    adj_idx[(size_t)row * KTOP + lane] = seli[lane];
    atomicAdd(&colcnt[seli[lane]], 1);
  }
  if (lane == 0) {
    float s = 1.f;
    for (int q = 0; q < KTOP; ++q) s += selv[q];
    rowsum[row] = s;
  }
}

// ---------------- small utility kernels ----------------------------------------------------
__global__ void k_zero(int* colcnt, int* cursor, float* tsum) {
  int i = blockIdx.x * blockDim.x + threadIdx.x;
  if (i < NN) { colcnt[i] = 0; cursor[i] = 0; tsum[i] = 0.f; }
}

__global__ __launch_bounds__(256) void k_prefix(const int* __restrict__ colcnt,
                                                int* __restrict__ colstart) {
  __shared__ int part[256];
  const int t = threadIdx.x;
  int s = 0;
  for (int i = 0; i < 20; ++i) { int g = t * 20 + i; if (g < NN) s += colcnt[g]; }
  part[t] = s;
  __syncthreads();
  if (t == 0) {
    int run = 0;
    for (int i = 0; i < 256; ++i) { int v = part[i]; part[i] = run; run += v; }
    colstart[NN] = run;   // == NNZ
  }
  __syncthreads();
  int run2 = part[t];
  for (int i = 0; i < 20; ++i) {
    int g = t * 20 + i;
    if (g < NN) { colstart[g] = run2; run2 += colcnt[g]; }
  }
}

__global__ void k_scatter(const float* __restrict__ adj_val, const int* __restrict__ adj_idx,
                          const int* __restrict__ colstart, int* __restrict__ cursor,
                          int* __restrict__ tidx, float* __restrict__ tval,
                          float* __restrict__ tsum) {
  int e = blockIdx.x * blockDim.x + threadIdx.x;
  if (e >= NNZ) return;
  int w = e / KTOP;
  float v = adj_val[e];
  int c = adj_idx[e];
  int pos = atomicAdd(&cursor[c], 1);
  int dst = colstart[c] + pos;
  tidx[dst] = w;
  tval[dst] = v;
  atomicAdd(&tsum[c], v);
}

// ---------------- kernel 4: dilated inception + gate, writes node-major xg -----------------
// block = 4 nodes x 48 (b,t) positions = 192 threads
__global__ __launch_bounds__(192) void k_inception(
    const float* __restrict__ x,
    const float* fw2, const float* fw3, const float* fw6, const float* fw7,
    const float* fb2, const float* fb3, const float* fb6, const float* fb7,
    const float* gw2, const float* gw3, const float* gw6, const float* gw7,
    const float* gb2, const float* gb3, const float* gb6, const float* gb7,
    float* __restrict__ xg) {
  __shared__ float wf[16][16][8];
  __shared__ float wg[16][16][8];
  __shared__ float bf[16], bg[16];
  const int tid = threadIdx.x;
  const float* wsrc[8] = {fw2, fw3, fw6, fw7, gw2, gw3, gw6, gw7};
  const float* bsrc[8] = {fb2, fb3, fb6, fb7, gb2, gb3, gb6, gb7};
  for (int e = tid; e < 2 * 16 * 16 * 8; e += 192) {
    int fg = e / 2048, rem = e % 2048;
    int c = rem / 128, i = (rem % 128) / 8, j = rem % 8;
    int br = c >> 2;
    int kb = (br == 0) ? 2 : (br == 1) ? 3 : (br == 2) ? 6 : 7;
    float v = 0.f;
    if (j < kb) v = wsrc[fg * 4 + br][((c & 3) * 16 + i) * kb + j];
    if (fg == 0) wf[c][i][j] = v; else wg[c][i][j] = v;
  }
  if (tid < 32) {
    int fg = tid / 16, c = tid % 16;
    float b = bsrc[fg * 4 + (c >> 2)][c & 3];
    if (fg == 0) bf[c] = b; else bg[c] = b;
  }
  __syncthreads();

  const int nl = tid / 48, p = tid % 48;
  const int b = p / TP, t = p % TP;
  const int n = blockIdx.x * 4 + nl;
  if (n >= NN) return;

  float accf[16], accg[16];
#pragma unroll
  for (int c = 0; c < 16; ++c) { accf[c] = bf[c]; accg[c] = bg[c]; }
  for (int i = 0; i < CIN; ++i) {
    const float* xp = x + (((size_t)b * CIN + i) * NN + n) * TT + t;
    float xv[7];
#pragma unroll
    for (int j = 0; j < 7; ++j) xv[j] = xp[j];
#pragma unroll
    for (int c = 0; c < 16; ++c) {
      const int kb = ((c >> 2) == 0) ? 2 : ((c >> 2) == 1) ? 3 : ((c >> 2) == 2) ? 6 : 7;
      float sf = 0.f, sg = 0.f;
#pragma unroll
      for (int j = 0; j < 7; ++j) {
        if (j < kb) { sf += wf[c][i][j] * xv[7 - kb + j]; sg += wg[c][i][j] * xv[7 - kb + j]; }
      }
      accf[c] += sf; accg[c] += sg;
    }
  }
#pragma unroll
  for (int c = 0; c < 16; ++c) {
    float f = fast_tanh(accf[c]);
    float g = fast_sigmoid(accg[c]);
    xg[(size_t)n * FEAT + ((size_t)b * CIN + c) * TP + t] = f * g;
  }
}

// ---------------- kernel 5: one mixprop propagation step (sparse gather SpMM) --------------
// mode 0: CSR (fixed 20 nnz/row, sums = rowsum incl. self). mode 1: CSC via seg, sum=1+tsum.
__global__ __launch_bounds__(256) void k_mp_step(const float* __restrict__ hin,
                                                 const float* __restrict__ xin,
                                                 float* __restrict__ hout,
                                                 const int* __restrict__ eidx,
                                                 const float* __restrict__ eval,
                                                 const int* __restrict__ seg,
                                                 const float* __restrict__ sums,
                                                 int mode) {
  const int v = blockIdx.x;
  const int tid = threadIdx.x;
  int s0, cnt; float sum;
  if (mode == 0) { s0 = v * KTOP; cnt = KTOP; sum = sums[v]; }
  else           { s0 = seg[v];   cnt = seg[v + 1] - s0; sum = 1.f + sums[v]; }
  const float inv = 1.f / sum;
  float a0 = 0.f, a1 = 0.f, a2 = 0.f;
  for (int e = 0; e < cnt; ++e) {
    int nb = eidx[s0 + e];
    float w = eval[s0 + e] * inv;
    const float* hp = hin + (size_t)nb * FEAT;
    a0 += w * hp[tid]; a1 += w * hp[tid + 256]; a2 += w * hp[tid + 512];
  }
  const float* sp = hin + (size_t)v * FEAT;
  a0 += inv * sp[tid]; a1 += inv * sp[tid + 256]; a2 += inv * sp[tid + 512];
  const float* xp = xin + (size_t)v * FEAT;
  float* op = hout + (size_t)v * FEAT;
  op[tid]       = ALPHA * xp[tid]       + (1.f - ALPHA) * a0;
  op[tid + 256] = ALPHA * xp[tid + 256] + (1.f - ALPHA) * a1;
  op[tid + 512] = ALPHA * xp[tid + 512] + (1.f - ALPHA) * a2;
}

// ---------------- kernel 6: mixprop output 1x1 conv over concat(x,h1,h2) -------------------
__global__ __launch_bounds__(256) void k_mp_out(const float* __restrict__ x,
                                                const float* __restrict__ h1,
                                                const float* __restrict__ h2,
                                                const float* __restrict__ W,
                                                const float* __restrict__ bias,
                                                float* __restrict__ out, int add) {
  __shared__ float sx[FEAT], s1[FEAT], s2[FEAT], sw[CIN * 48];
  __shared__ float sb[CIN];
  const int n = blockIdx.x;
  const int tid = threadIdx.x;
  for (int e = tid; e < FEAT; e += 256) {
    sx[e] = x[(size_t)n * FEAT + e];
    s1[e] = h1[(size_t)n * FEAT + e];
    s2[e] = h2[(size_t)n * FEAT + e];
    sw[e] = W[e];
  }
  if (tid < CIN) sb[tid] = bias[tid];
  __syncthreads();
#pragma unroll
  for (int q = 0; q < 3; ++q) {
    int p = tid + 256 * q;
    int b = p / (CIN * TP), o = (p / TP) % CIN, t = p % TP;
    float acc = sb[o];
    int fb = (b * CIN) * TP + t;
#pragma unroll
    for (int c = 0; c < CIN; ++c) {
      int f = fb + c * TP;
      acc += sw[o * 48 + c]      * sx[f]
           + sw[o * 48 + 16 + c] * s1[f]
           + sw[o * 48 + 32 + c] * s2[f];
    }
    size_t oi = (((size_t)b * CIN + o) * NN + n) * TP + t;
    if (add) out[oi] += acc; else out[oi] = acc;
  }
}

// ---------------- host-side launch ---------------------------------------------------------
extern "C" void kernel_launch(void* const* d_in, const int* in_sizes, int n_in,
                              void* d_out, int out_size, void* d_ws, size_t ws_size,
                              hipStream_t stream) {
  const float* x      = (const float*)d_in[0];
  const int*   idxmap = (const int*)  d_in[1];
  const float* emb1   = (const float*)d_in[2];
  const float* emb2   = (const float*)d_in[3];
  const float* l1w    = (const float*)d_in[4];
  const float* l1b    = (const float*)d_in[5];
  const float* l2w    = (const float*)d_in[6];
  const float* l2b    = (const float*)d_in[7];
  const float* fw2 = (const float*)d_in[8],  *fb2 = (const float*)d_in[9];
  const float* fw3 = (const float*)d_in[10], *fb3 = (const float*)d_in[11];
  const float* fw6 = (const float*)d_in[12], *fb6 = (const float*)d_in[13];
  const float* fw7 = (const float*)d_in[14], *fb7 = (const float*)d_in[15];
  const float* gw2 = (const float*)d_in[16], *gb2 = (const float*)d_in[17];
  const float* gw3 = (const float*)d_in[18], *gb3 = (const float*)d_in[19];
  const float* gw6 = (const float*)d_in[20], *gb6 = (const float*)d_in[21];
  const float* gw7 = (const float*)d_in[22], *gb7 = (const float*)d_in[23];
  const float* mp1w = (const float*)d_in[24], *mp1b = (const float*)d_in[25];
  const float* mp2w = (const float*)d_in[26], *mp2b = (const float*)d_in[27];
  float* out = (float*)d_out;

  // workspace carve (256B aligned slots)
  size_t off = 0;
  auto carve = [&](size_t bytes) -> void* {
    void* p = (char*)d_ws + off;
    off = (off + bytes + 255) & ~(size_t)255;
    return p;
  };
  _Float16* nv1 = (_Float16*)carve((size_t)NP * KP * 2);
  _Float16* nv2 = (_Float16*)carve((size_t)NP * KP * 2);
  float* cand_val = (float*)carve((size_t)NN * NSPLIT * KTOP * 4);
  int*   cand_idx = (int*)  carve((size_t)NN * NSPLIT * KTOP * 4);
  float* adj_val  = (float*)carve((size_t)NNZ * 4);
  int*   adj_idx  = (int*)  carve((size_t)NNZ * 4);
  float* rowsum   = (float*)carve((size_t)NN * 4);
  int*   colcnt   = (int*)  carve((size_t)NN * 4);
  int*   colstart = (int*)  carve((size_t)(NN + 1) * 4);
  int*   cursor   = (int*)  carve((size_t)NN * 4);
  int*   tidx     = (int*)  carve((size_t)NNZ * 4);
  float* tval     = (float*)carve((size_t)NNZ * 4);
  float* tsum     = (float*)carve((size_t)NN * 4);
  float* xg       = (float*)carve((size_t)NN * FEAT * 4);
  float* h1       = (float*)carve((size_t)NN * FEAT * 4);
  float* h2       = (float*)carve((size_t)NN * FEAT * 4);
  (void)ws_size; (void)n_in; (void)in_sizes; (void)out_size;

  // 1) node vectors (f16, K/N padded)
  k_node_vecs<<<NP, 64, 0, stream>>>(emb1, idxmap, l1w, l1b, nv1);
  k_node_vecs<<<NP, 64, 0, stream>>>(emb2, idxmap, l2w, l2b, nv2);

  // 2) zero counters (ws is poisoned each run)
  k_zero<<<(NN + 255) / 256, 256, 0, stream>>>(colcnt, cursor, tsum);

  // 3) WMMA similarity (a = nv1.nv2^T - nv2.nv1^T) fused with per-row top-20 on raw values
  k_simtopk<<<dim3(NT, NSPLIT), 32, 0, stream>>>(nv1, nv2, cand_val, cand_idx);

  // 4) merge split candidates -> CSR top-20 (deferred tanh), row sums, column counts
  k_merge<<<NN, 32, 0, stream>>>(cand_val, cand_idx, adj_val, adj_idx, rowsum, colcnt);

  // 5) CSC build for the transposed-graph mixprop
  k_prefix<<<1, 256, 0, stream>>>(colcnt, colstart);
  k_scatter<<<(NNZ + 255) / 256, 256, 0, stream>>>(adj_val, adj_idx, colstart, cursor,
                                                   tidx, tval, tsum);

  // 6) dilated inception + gate -> node-major features xg [N,768]
  k_inception<<<(NN + 3) / 4, 192, 0, stream>>>(x,
      fw2, fw3, fw6, fw7, fb2, fb3, fb6, fb7,
      gw2, gw3, gw6, gw7, gb2, gb3, gb6, gb7, xg);

  // 7) mixprop 1 (adj): two propagation steps + 1x1 conv, writes out
  k_mp_step<<<NN, 256, 0, stream>>>(xg, xg, h1, adj_idx, adj_val, nullptr, rowsum, 0);
  k_mp_step<<<NN, 256, 0, stream>>>(h1, xg, h2, adj_idx, adj_val, nullptr, rowsum, 0);
  k_mp_out<<<NN, 256, 0, stream>>>(xg, h1, h2, mp1w, mp1b, out, 0);

  // 8) mixprop 2 (adj^T via CSC): accumulates into out
  k_mp_step<<<NN, 256, 0, stream>>>(xg, xg, h1, tidx, tval, colstart, tsum, 1);
  k_mp_step<<<NN, 256, 0, stream>>>(h1, xg, h2, tidx, tval, colstart, tsum, 1);
  k_mp_out<<<NN, 256, 0, stream>>>(xg, h1, h2, mp2w, mp2b, out, 1);
}
